// gConv3d_61761629716539
// MI455X (gfx1250) — matrix operand order repaired and checked by reading, not compiled, not run on hardware
//
#include <hip/hip_runtime.h>

typedef __attribute__((ext_vector_type(16))) _Float16 v16h;
typedef __attribute__((ext_vector_type(8)))  _Float16 v8h;
typedef __attribute__((ext_vector_type(8)))  float    v8f;

// Problem constants
#define CC   8     // input channels
#define OO   8
#define GG   12
#define GO_  96    // G*O
#define XD   12
#define CH_  8
#define CW_  40
#define XO_  9     // X-3
#define NN   729   // XO^3
#define PH   6     // CH-2
#define PW   38    // CW-2
#define NCOL 228   // PH*PW
#define KO_N 9     // (fi,fj) outer blocks
#define KS_N 7     // 32-wide k-steps per outer block (224 padded)
#define KIN  216   // real inner K per outer block: (fk*9 + u*3+v)*8 + c
#define AFRAG_HALFS (6 * KO_N * KS_N * 512)   // ko * mt * ks * (32 lanes * 16 halfs)
#define A_KO_HALFS  (6 * KS_N * 512)          // 21504 halfs per ko chunk
#define A_KO_CHUNKS (A_KO_HALFS / 8)          // 2688 uint4 chunks

// x strides (B=1): (C, X, Y, Z, H, W)
#define XS_C 552960  // 12*12*12*8*40
#define XS_X 46080   // 12*12*8*40
#define XS_Y 3840    // 12*8*40
#define XS_Z 320     // 8*40

// ws layout (bytes)
#define BIAS_OFF 387072   // = AFRAG_HALFS*2
#define ACC_OFF  393216

// Compile-time base (half units) of an 8-channel K group in the LDS slab.
// Slab layout: [fk][h][w][c] halfs; group g = fk*9 + (u*3+v), g in [0,27).
__device__ constexpr int groupBase(int g) {
  if (g >= 27) return -1;        // pad group (kLocal 216..223)
  int fk = g / 9;
  int uv = g - 9 * fk;
  int u  = uv / 3;
  int v  = uv - 3 * u;
  return ((fk * 8 + u) * 40 + v) * 8;
}

// ---------------------------------------------------------------------------
// 1) Build A fragments (f16, WMMA A layout, K order (fk,u,v,c)) + biasSum
//    afrag layout: [ko][mt][ks][lane][16h]  (per-ko chunk contiguous)
// ---------------------------------------------------------------------------
__global__ void prep_kernel(const float* __restrict__ weight,
                            const float* __restrict__ bias,
                            const float* __restrict__ basis,
                            _Float16* __restrict__ afrag,
                            float* __restrict__ biasSum) {
  int t = blockIdx.x * 256 + threadIdx.x;
  if (t < OO) {
    float s = 0.f;
    for (int i = 0; i < 27; ++i) s += bias[t * 27 + i];
    biasSum[t] = s;
  }
  if (t >= AFRAG_HALFS) return;
  int h    = t & 15;
  int lane = (t >> 4) & 31;
  int rest = t >> 9;                 // (ko*6 + mt)*7 + ks
  int ks = rest % KS_N; rest /= KS_N;
  int mt = rest % 6;
  int ko = rest / 6;
  // 16-bit A-matrix VGPR layout: lane<16 -> K {0..7,16..23}; lane>=16 -> +8
  int kOff   = (h & 7) + ((h >> 3) << 4) + (((lane >> 4) & 1) << 3);
  int kLocal = ks * 32 + kOff;
  float val = 0.f;
  if (kLocal < KIN) {
    int c   = kLocal & 7;          // channel innermost
    int grp = kLocal >> 3;
    int fk  = grp / 9;
    int uv  = grp - 9 * fk;
    int u   = uv / 3;
    int v   = uv - 3 * u;
    int fi = ko / 3, fj = ko - 3 * (ko / 3);
    int M = mt * 16 + (lane & 15);       // go index, < 96
    int g = M >> 3;                      // O == 8
    int o = M & 7;
    const float* wp = weight + ((((o * CC + c) * 3 + fi) * 3 + fj) * 3 + fk) * 7;
    const float* bp = basis + g * 63 + u * 3 + v;   // basis[g,b,u,v], b-stride 9
    float s = 0.f;
    for (int b = 0; b < 7; ++b) s += wp[b] * bp[b * 9];
    val = s;
  }
  afrag[t] = (_Float16)val;
}

// ---------------------------------------------------------------------------
// 2) Implicit-GEMM conv: one block per macro-position n; 8 wave32s.
// ---------------------------------------------------------------------------
__global__ void __launch_bounds__(256) gemm_kernel(const float* __restrict__ x,
                                                   const _Float16* __restrict__ afrag,
                                                   float* __restrict__ acc_out) {
  __shared__ __align__(16) _Float16 slab[3 * 8 * 40 * 8];   // [fk][h][w][c], 15360 B
  __shared__ __align__(16) _Float16 aslab[A_KO_HALFS];      // per-ko A chunk, 43008 B
  const int n   = blockIdx.x;
  const int zo  = n % 9;
  const int yo  = (n / 9) % 9;
  const int xo  = n / 81;
  const int tid  = threadIdx.x;
  const int lane = tid & 31;
  const int wave = tid >> 5;          // 0..7
  const int nt0  = wave * 2;          // this wave's two column tiles
  const bool hiHalf = lane >= 16;

  int col[2], pqOff[2];
#pragma unroll
  for (int j = 0; j < 2; ++j) {
    int c0 = (nt0 + j) * 16 + (lane & 15);
    col[j] = c0;
    int cc = c0 > (NCOL - 1) ? (NCOL - 1) : c0;   // clamp pad columns (finite garbage)
    int p = cc / PW;
    int q = cc - PW * p;
    pqOff[j] = (p * CW_ + q) * 8;    // half units, 16B aligned
  }

  const v8f zero = {0.f, 0.f, 0.f, 0.f, 0.f, 0.f, 0.f, 0.f};
  v8f acc[2][6];
#pragma unroll
  for (int j = 0; j < 2; ++j)
#pragma unroll
    for (int mt = 0; mt < 6; ++mt) acc[j][mt] = zero;

  for (int ko = 0; ko < KO_N; ++ko) {
    const int fi = ko / 3, fj = ko - 3 * (ko / 3);
    __syncthreads();
    // (a) Stage x-slab as f16, channel-interleaved [fk][h][w][c].
    const float* xb = x + (xo + fi) * XS_X + (yo + fj) * XS_Y;
#pragma unroll
    for (int i = 0; i < 8; ++i) {
      int sc = tid + 256 * i;
      if (sc < 1920) {
        int wq = sc % 10;
        int t1 = sc / 10;
        int hh = t1 & 7;
        int t2 = t1 >> 3;
        int c  = t2 & 7;
        int fk = t2 >> 3;
        int ww = wq * 4;
        const float4 xv = *(const float4*)(xb + c * XS_C + (zo + fk) * XS_Z + hh * 40 + ww);
        int d = ((fk * 8 + hh) * 40 + ww) * 8 + c;
        slab[d]      = (_Float16)xv.x;
        slab[d + 8]  = (_Float16)xv.y;
        slab[d + 16] = (_Float16)xv.z;
        slab[d + 24] = (_Float16)xv.w;
      }
    }
    // (b) Stage this ko's A chunk (contiguous) into LDS: 2688 uint4 copies.
    {
      const uint4* asrc = (const uint4*)afrag + ko * A_KO_CHUNKS;
      uint4* adst = (uint4*)aslab;
#pragma unroll
      for (int i = 0; i < 11; ++i) {
        int idx = tid + 256 * i;
        if (idx < A_KO_CHUNKS) adst[idx] = asrc[idx];
      }
    }
    __syncthreads();

#pragma unroll
    for (int ks = 0; ks < KS_N; ++ks) {
      // B fragment = two ds_load_b128 per column tile: halfs 0..7 are the
      // 8 channels of group (ks*4 [+1 hi-half]); halfs 8..15 are group +2.
      const int gLo   = ks * 4;
      const int bA_lo = groupBase(gLo);
      const int bA_hi = groupBase(gLo + 1);
      const int bB_lo = groupBase(gLo + 2);
      const int bB_hi = groupBase(gLo + 3);   // -1 only at ks==6
      const int bA = hiHalf ? bA_hi : bA_lo;  // one v_cndmask between constants

      v16h bfr[2];
#pragma unroll
      for (int j = 0; j < 2; ++j) {
        v8h lo = *(const v8h*)(slab + bA + pqOff[j]);
        v8h hi;
        if (bB_hi >= 0) {
          int bB = hiHalf ? bB_hi : bB_lo;
          hi = *(const v8h*)(slab + bB + pqOff[j]);
        } else {
          v8h tpad = *(const v8h*)(slab + bB_lo + pqOff[j]);
          v8h z8 = {0, 0, 0, 0, 0, 0, 0, 0};
          hi = hiHalf ? z8 : tpad;            // hi lanes: zero pad group
        }
#pragma unroll
        for (int e = 0; e < 8; ++e) {
          bfr[j][e]     = lo[e];
          bfr[j][8 + e] = hi[e];
        }
      }

      // Preload all 6 A fragments from LDS (short-latency ds_load_b128),
      // then 12 back-to-back WMMAs.
      const _Float16* abase = aslab + ks * 512 + lane * 16;
      v16h a[6];
#pragma unroll
      for (int mt = 0; mt < 6; ++mt)
        a[mt] = *(const v16h*)(abase + mt * (KS_N * 512));
#pragma unroll
      for (int mt = 0; mt < 6; ++mt) {
        acc[0][mt] = __builtin_amdgcn_wmma_f32_16x16x32_f16(
            false, a[mt], false, bfr[0], (short)0, acc[0][mt], false, false);
        acc[1][mt] = __builtin_amdgcn_wmma_f32_16x16x32_f16(
            false, a[mt], false, bfr[1], (short)0, acc[1][mt], false, false);
      }
    }
  }

  // Epilogue: C/D layout — row M = r + 8*(lane>=16), col N = lane&15
  const int rowAdd = (lane >> 4) << 3;
#pragma unroll
  for (int j = 0; j < 2; ++j) {
    if (col[j] < NCOL) {
      int p = col[j] / PW;
      int q = col[j] - PW * p;
#pragma unroll
      for (int mt = 0; mt < 6; ++mt)
#pragma unroll
        for (int r = 0; r < 8; ++r) {
          int go = mt * 16 + r + rowAdd;
          acc_out[((n * GO_ + go) * PH + p) * PW + q] = acc[j][mt][r];
        }
    }
  }
}

// ---------------------------------------------------------------------------
// 3) Gather/permute epilogue + zero padding; float4 (16B) coalesced stores
// ---------------------------------------------------------------------------
__global__ void scatter_kernel(const float* __restrict__ acc,
                               const float* __restrict__ biasSum,
                               const int* __restrict__ I,
                               const int* __restrict__ J,
                               const int* __restrict__ T,
                               const int* __restrict__ bias_basis,
                               float* __restrict__ out, int total4) {
  int qidx = blockIdx.x * 256 + threadIdx.x;
  if (qidx >= total4) return;
  int idx = qidx * 4;
  int w0 = idx % CW_;  int t = idx / CW_;   // w0 multiple of 4
  int h = t % CH_;     t /= CH_;
  int z = t % XD;      t /= XD;
  int y = t % XD;      t /= XD;
  int xx = t % XD;
  int go = t / XD;
  float4 val = {0.f, 0.f, 0.f, 0.f};
  if (xx >= 1 && xx <= XO_ && y >= 1 && y <= XO_ && z >= 1 && z <= XO_) {
    int n = ((xx - 1) * 9 + (y - 1)) * 9 + (z - 1);
    int g = go >> 3;
    int o = go & 7;
    float bs = biasSum[bias_basis[go]];
    float* ve = &val.x;
#pragma unroll
    for (int e = 0; e < 4; ++e) {
      int w  = w0 + e;
      int gg = T[(g * CH_ + h) * CW_ + w];
      int hh = I[h * CW_ + w] - 1;
      int ww = J[h * CW_ + w] - 1;
      ve[e] = acc[((n * GO_ + gg * OO + o) * PH + hh) * PW + ww] + bs;
    }
  }
  *(float4*)(out + idx) = val;
}

// ---------------------------------------------------------------------------
extern "C" void kernel_launch(void* const* d_in, const int* in_sizes, int n_in,
                              void* d_out, int out_size, void* d_ws, size_t ws_size,
                              hipStream_t stream) {
  const float* x      = (const float*)d_in[0];
  const float* weight = (const float*)d_in[1];
  const float* bias   = (const float*)d_in[2];
  const float* basis  = (const float*)d_in[3];
  const int*   I      = (const int*)d_in[4];
  const int*   J      = (const int*)d_in[5];
  const int*   T      = (const int*)d_in[6];
  const int*   bb     = (const int*)d_in[7];

  char*     ws      = (char*)d_ws;
  _Float16* afrag   = (_Float16*)ws;
  float*    biasSum = (float*)(ws + BIAS_OFF);
  float*    acc     = (float*)(ws + ACC_OFF);

  prep_kernel<<<AFRAG_HALFS / 256, 256, 0, stream>>>(weight, bias, basis, afrag, biasSum);
  gemm_kernel<<<NN, 256, 0, stream>>>(x, afrag, acc);

  int total4 = out_size / 4;
  scatter_kernel<<<(total4 + 255) / 256, 256, 0, stream>>>(acc, biasSum, I, J, T, bb,
                                                           (float*)d_out, total4);
}
